// QLSTMHybrid_65481071396976
// MI455X (gfx1250) — matrix operand order repaired and checked by reading, not compiled, not run on hardware
//
#include <hip/hip_runtime.h>

// ---------------------------------------------------------------------------
// Types for CDNA5 WMMA (wave32): v_wmma_f32_16x16x32_bf16
// ---------------------------------------------------------------------------
typedef __attribute__((ext_vector_type(16))) __bf16 v16bf;
typedef __attribute__((ext_vector_type(8)))  float  v8f;

#define T_STEPS 4096
#define E_DIM   1024
#define H_DIM   1024
#define G4      4096          // 4*H
#define TAGS_N  32000
#define LSTM_WG 32

// ---------------------------------------------------------------------------
// CDNA5 async global->LDS copy (ISA §15.18.3 op 98, VGLOBAL async encoding):
//   GLOBAL_LOAD_ASYNC_TO_LDS_B128  vdst(=LDS byte addr), vaddr(64b), off
// Tracked by ASYNCcnt; drained with S_WAIT_ASYNCCNT (SOPP op 74).
// The clang builtin is absent on this toolchain (probe round 3), so inline asm.
// LDS operand = low 32 bits of the generic pointer.
// ---------------------------------------------------------------------------
__device__ __forceinline__ void async_load_b128_to_lds(const void* gptr,
                                                       void* lptr) {
    unsigned int loff = (unsigned int)(unsigned long long)lptr;
    asm volatile("global_load_async_to_lds_b128 %0, %1, off"
                 :
                 : "v"(loff), "v"(gptr)
                 : "memory");
}
__device__ __forceinline__ void wait_async_lds() {
    asm volatile("s_wait_asynccnt 0" ::: "memory");
}

__device__ __forceinline__ unsigned short f2bf(float f) {
    unsigned int u = __builtin_bit_cast(unsigned int, f);
    unsigned int r = u + 0x7FFFu + ((u >> 16) & 1u);   // round-to-nearest-even
    return (unsigned short)(r >> 16);
}
__device__ __forceinline__ float bf2f(unsigned short h) {
    unsigned int u = ((unsigned int)h) << 16;
    return __builtin_bit_cast(float, u);
}
__device__ __forceinline__ float sigmoid_f(float x) {
    return 1.f / (1.f + __expf(-x));
}
__device__ __forceinline__ float tanh_f(float x) {
    float e2 = __expf(2.f * x);
    return (e2 - 1.f) / (e2 + 1.f);
}

// ---------------------------------------------------------------------------
// Init: zero h state and grid-barrier counter (ws is poisoned 0xAA once).
// ---------------------------------------------------------------------------
__global__ void k_init(float* hbuf, unsigned int* bar) {
    for (int i = threadIdx.x; i < H_DIM; i += 256) hbuf[i] = 0.f;
    if (threadIdx.x == 0) *bar = 0u;
}

// ---------------------------------------------------------------------------
// Prepare: W4B (first E rows of [Wf|Wi|Wg|Wo], bf16, [1024][4096])
// ---------------------------------------------------------------------------
__global__ void k_build_w4b(const float* Wf, const float* Wi, const float* Wg,
                            const float* Wo, unsigned short* W4B) {
    int idx = blockIdx.x * 256 + threadIdx.x;          // [0, 1024*4096)
    int r = idx >> 12, c = idx & 4095;
    int gate = c >> 10, cl = c & 1023;
    const float* W = (gate == 0) ? Wf : (gate == 1) ? Wi : (gate == 2) ? Wg : Wo;
    W4B[idx] = f2bf(W[(size_t)r * H_DIM + cl]);
}

// Wh transposed: Wh4T[col][h] = W4[E+h][col], bf16, [4096][1024]
__global__ void k_build_wh4t(const float* Wf, const float* Wi, const float* Wg,
                             const float* Wo, unsigned short* Wh4T) {
    int idx = blockIdx.x * 256 + threadIdx.x;          // [0, 4096*1024)
    int col = idx >> 10, h = idx & 1023;
    int gate = col >> 10, cl = col & 1023;
    const float* W = (gate == 0) ? Wf : (gate == 1) ? Wi : (gate == 2) ? Wg : Wo;
    Wh4T[idx] = f2bf(W[(size_t)(E_DIM + h) * H_DIM + cl]);
}

__global__ void k_build_b4(const float* bf_, const float* bi, const float* bg,
                           const float* bo, float* b4) {
    int c = blockIdx.x * 256 + threadIdx.x;            // [0, 4096)
    int gate = c >> 10, cl = c & 1023;
    const float* b = (gate == 0) ? bf_ : (gate == 1) ? bi : (gate == 2) ? bg : bo;
    b4[c] = b[cl];
}

// Embedding gather + bf16 convert: embB[t][e] = bf16(emb[sentence[t]][e])
__global__ void k_gather_emb(const int* sentence, const float* emb,
                             unsigned short* embB) {
    int idx = blockIdx.x * 256 + threadIdx.x;          // [0, 4096*1024)
    int t = idx >> 10, e = idx & 1023;
    int row = sentence[t];
    embB[idx] = f2bf(emb[(size_t)row * E_DIM + e]);
}

// Plain f32 -> bf16 convert (for W_tag)
__global__ void k_conv_bf16(const float* in, unsigned short* out) {
    int idx = blockIdx.x * 256 + threadIdx.x;
    out[idx] = f2bf(in[idx]);
}

// ---------------------------------------------------------------------------
// bf16 WMMA GEMM: C[M,N] = A[M,K] * B[K,N] + bias[N]
// A,B row-major bf16 (raw u16); C f32. M%128==0, N%64==0, K%32==0.
// Block = 256 threads (8 waves), 128x64 C tile; each wave 32x32 = 4 WMMAs/step.
// Double-buffered LDS: stage tile k+1 (async global->LDS A copy + transposed
// B stores) overlapped with the 4 WMMAs on tile k; one barrier per K-step.
// ---------------------------------------------------------------------------
__global__ __launch_bounds__(256)
void k_gemm_bf16(const unsigned short* __restrict__ A,
                 const unsigned short* __restrict__ B,
                 float* __restrict__ C, const float* __restrict__ bias,
                 int K, int lda, int ldb, int ldc) {
    __shared__ unsigned short At[2][128 * 32];         // A tiles row-major (16 KB)
    __shared__ unsigned short Bt[2][64 * 36];          // B tiles transposed (9 KB)

    const int tid  = threadIdx.x;
    const int lane = tid & 31;
    const int wid  = tid >> 5;
    const int wm   = wid & 3;                          // 4 M sub-tiles of 32
    const int wn   = wid >> 2;                         // 2 N sub-tiles of 32
    const int m0   = blockIdx.y * 128;
    const int n0   = blockIdx.x * 64;

    const int koffA = (lane < 16) ? 0 : 16;            // A-frag K offset (contiguous K)
    const int koffB = (lane < 16) ? 0 : 16;            // B-frag K offset

    const int sr = tid >> 3;                           // B staging: row 0..31
    const int sc = (tid & 7) * 8;                      // B staging: col group

    v8f acc00 = {}, acc01 = {}, acc10 = {}, acc11 = {};

    auto stage = [&](int k0, int buf) {
        // B[k0..k0+32) x [n0..n0+64) -> LDS transposed (needs VGPR transpose)
        uint4 bq = *(const uint4*)(B + (size_t)(k0 + sr) * ldb + n0 + sc);
        const unsigned short* bs = (const unsigned short*)&bq;
#pragma unroll
        for (int e = 0; e < 8; ++e)
            Bt[buf][(sc + e) * 36 + sr] = bs[e];
        // A[m0..m0+128) x [k0..k0+32) -> LDS via async DMA (no transpose needed)
#pragma unroll
        for (int i = 0; i < 2; ++i) {
            const int lin = tid + i * 256;             // [0, 512)
            const int r   = lin >> 2;                  // 0..127
            const int cg  = (lin & 3) * 8;             // 0,8,16,24
            async_load_b128_to_lds(A + (size_t)(m0 + r) * lda + k0 + cg,
                                   &At[buf][r * 32 + cg]);
        }
        // speculative prefetch of the B tile after next (dropped if OOB)
        __builtin_prefetch(B + (size_t)(k0 + 32 + sr) * ldb + n0 + sc, 0, 1);
    };

    stage(0, 0);
    const int KT = K >> 5;                             // K / 32 tiles
    for (int kt = 0; kt < KT; ++kt) {
        const int cur = kt & 1;
        wait_async_lds();                              // tile kt landed in LDS
        __syncthreads();                               // ...and prev compute done
        if (kt + 1 < KT)                               // overlap stage(k+1) w/ WMMA
            stage((kt + 1) << 5, cur ^ 1);

        const unsigned short* Atc = At[cur];
        const unsigned short* Btc = Bt[cur];

        // --- A fragments: 16x32 bf16 each, two b128 LDS loads per lane ---
        union { v16bf v; uint4 q[2]; } af0, af1;
        const int mr = wm * 32 + (lane & 15);
        af0.q[0] = *(const uint4*)&Atc[(mr)      * 32 + koffA];
        af0.q[1] = *(const uint4*)&Atc[(mr)      * 32 + 8 + koffA];
        af1.q[0] = *(const uint4*)&Atc[(mr + 16) * 32 + koffA];
        af1.q[1] = *(const uint4*)&Atc[(mr + 16) * 32 + 8 + koffA];

        // --- B fragments from LDS (contiguous K pairs per VGPR) ---
        union { v16bf v; unsigned int u[8]; } bf0, bf1;
        const int nl0 = wn * 32 + (lane & 15);
#pragma unroll
        for (int vv = 0; vv < 8; ++vv) {
            bf0.u[vv] = *(const unsigned int*)&Btc[nl0 * 36 + 2 * vv + koffB];
            bf1.u[vv] = *(const unsigned int*)&Btc[(nl0 + 16) * 36 + 2 * vv + koffB];
        }

        acc00 = __builtin_amdgcn_wmma_f32_16x16x32_bf16(false, af0.v, false, bf0.v,
                                                        (short)0, acc00, false, false);
        acc01 = __builtin_amdgcn_wmma_f32_16x16x32_bf16(false, af0.v, false, bf1.v,
                                                        (short)0, acc01, false, false);
        acc10 = __builtin_amdgcn_wmma_f32_16x16x32_bf16(false, af1.v, false, bf0.v,
                                                        (short)0, acc10, false, false);
        acc11 = __builtin_amdgcn_wmma_f32_16x16x32_bf16(false, af1.v, false, bf1.v,
                                                        (short)0, acc11, false, false);
    }

    // C/D layout: VGPR r -> M = r + (lane<16 ? 0 : 8); N = lane&15
    const int mb0 = m0 + wm * 32 + ((lane < 16) ? 0 : 8);
    const int nA  = n0 + wn * 32 + (lane & 15);
    const int nB  = nA + 16;
    const float bvA = bias[nA];
    const float bvB = bias[nB];
#pragma unroll
    for (int r = 0; r < 8; ++r) {
        C[(size_t)(mb0 + r) * ldc + nA]      = acc00[r] + bvA;
        C[(size_t)(mb0 + r) * ldc + nB]      = acc01[r] + bvB;
        C[(size_t)(mb0 + 16 + r) * ldc + nA] = acc10[r] + bvA;
        C[(size_t)(mb0 + 16 + r) * ldc + nB] = acc11[r] + bvB;
    }
}

// ---------------------------------------------------------------------------
// Sequential LSTM recurrence. 32 persistent WGs x 256 threads.
// Block b owns j in [32b, 32b+32). Per step: z = X4[t] (has x-part + b4)
//                                             + hprev @ Wh4T-cols.
// Grid sync via monotonic device-scope atomic counter.
// ---------------------------------------------------------------------------
__global__ __launch_bounds__(256)
void k_lstm(const unsigned short* __restrict__ Wh4T,   // [4096][1024] bf16
            const float* __restrict__ X4,              // [T][4096], incl. b4
            float* __restrict__ hbuf,                  // [1024]
            unsigned short* __restrict__ hseqB,        // [T][1024] bf16
            unsigned int* __restrict__ bar) {
    __shared__ float hprev[H_DIM];
    __shared__ float part[256];
    __shared__ float gates[128];
    __shared__ float cst[32];

    const int tid  = threadIdx.x;
    const int blk  = blockIdx.x;
    const int out  = tid & 127;            // gate*32 + j_local
    const int half = tid >> 7;             // which 512-wide K half
    const int gate = out >> 5;
    const int jl   = out & 31;
    const int j    = blk * 32 + jl;
    const int col  = gate * H_DIM + j;
    const uint4* wp = (const uint4*)(Wh4T + (size_t)col * H_DIM + half * 512);

    if (tid < 32) cst[tid] = 0.f;

    for (int t = 0; t < T_STEPS; ++t) {
        if (t == 0) {
            for (int i = tid; i < H_DIM; i += 256) hprev[i] = 0.f;
        } else {
            for (int i = tid; i < H_DIM; i += 256)
                hprev[i] = __hip_atomic_load(&hbuf[i], __ATOMIC_RELAXED,
                                             __HIP_MEMORY_SCOPE_AGENT);
        }
        __syncthreads();

        // 512-long partial dot: contiguous b128 weight stream (L2 resident)
        float s = 0.f;
        const float* hp = &hprev[half * 512];
#pragma unroll 4
        for (int i = 0; i < 64; ++i) {
            uint4 w = wp[i];
            const unsigned short* ws = (const unsigned short*)&w;
#pragma unroll
            for (int e = 0; e < 8; ++e)
                s = fmaf(bf2f(ws[e]), hp[i * 8 + e], s);
        }
        part[tid] = s;
        __syncthreads();

        if (tid < 128) {
            float z = part[tid] + part[tid + 128] + X4[(size_t)t * G4 + col];
            gates[out] = (gate == 2) ? tanh_f(z) : sigmoid_f(z);
        }
        __syncthreads();

        if (tid < 32) {
            float f  = gates[tid];
            float ii = gates[32 + tid];
            float g  = gates[64 + tid];
            float o  = gates[96 + tid];
            float c  = f * cst[tid] + ii * g;
            cst[tid] = c;
            float h = o * tanh_f(c);
            __hip_atomic_store(&hbuf[blk * 32 + tid], h, __ATOMIC_RELAXED,
                               __HIP_MEMORY_SCOPE_AGENT);
            hseqB[(size_t)t * H_DIM + blk * 32 + tid] = f2bf(h);
        }
        __syncthreads();

        // grid barrier: monotonic counter, release on arrive / acquire on spin
        if (tid == 0) {
            __hip_atomic_fetch_add(bar, 1u, __ATOMIC_ACQ_REL,
                                   __HIP_MEMORY_SCOPE_AGENT);
            const unsigned int target = (unsigned int)(t + 1) * LSTM_WG;
            while (__hip_atomic_load(bar, __ATOMIC_ACQUIRE,
                                     __HIP_MEMORY_SCOPE_AGENT) < target)
                __builtin_amdgcn_s_sleep(1);
        }
        __syncthreads();
    }
}

// ---------------------------------------------------------------------------
// Sampler stage 1: s1[t][k] = tanh(sum_j logits[t][j] * Ws1[j][k] + bs1[k])
// One block per t row; logits live in d_out (scratch reuse).
// ---------------------------------------------------------------------------
__global__ __launch_bounds__(256)
void k_samp1(const float* __restrict__ logits, const float* __restrict__ Ws1,
             const float* __restrict__ bs1, float* __restrict__ s1) {
    __shared__ float4 red[256];
    const int t = blockIdx.x, tid = threadIdx.x;
    float a0 = 0, a1 = 0, a2 = 0, a3 = 0;
    const float* lp = logits + (size_t)t * TAGS_N;
    for (int jj = tid; jj < TAGS_N; jj += 256) {
        float x = lp[jj];
        const float* w = Ws1 + (size_t)jj * 4;
        a0 = fmaf(x, w[0], a0); a1 = fmaf(x, w[1], a1);
        a2 = fmaf(x, w[2], a2); a3 = fmaf(x, w[3], a3);
    }
    red[tid] = make_float4(a0, a1, a2, a3);
    __syncthreads();
    for (int s = 128; s > 0; s >>= 1) {
        if (tid < s) {
            red[tid].x += red[tid + s].x; red[tid].y += red[tid + s].y;
            red[tid].z += red[tid + s].z; red[tid].w += red[tid + s].w;
        }
        __syncthreads();
    }
    if (tid < 4) {
        float v = (&red[0].x)[tid] + bs1[tid];
        s1[(size_t)t * 4 + tid] = tanh_f(v);
    }
}

// ---------------------------------------------------------------------------
// Sampler stage 2 + log-softmax: z = s1 @ Ws2 + bs2; out = log(softmax(z)+1e-12)
// z recomputed per pass (4 FMAs/elem; Ws2 rows cached) — saves 512MB scratch.
// ---------------------------------------------------------------------------
__global__ __launch_bounds__(256)
void k_samp2(const float* __restrict__ s1, const float* __restrict__ Ws2,
             const float* __restrict__ bs2, float* __restrict__ outp) {
    __shared__ float red[256];
    const int t = blockIdx.x, tid = threadIdx.x;
    const float s0 = s1[(size_t)t * 4 + 0], s1v = s1[(size_t)t * 4 + 1];
    const float s2 = s1[(size_t)t * 4 + 2], s3v = s1[(size_t)t * 4 + 3];

    // pass 1: row max
    float m = -3.4e38f;
    for (int jj = tid; jj < TAGS_N; jj += 256) {
        float z = fmaf(s0, Ws2[jj], fmaf(s1v, Ws2[TAGS_N + jj],
                  fmaf(s2, Ws2[2 * TAGS_N + jj],
                  fmaf(s3v, Ws2[3 * TAGS_N + jj], bs2[jj]))));
        m = fmaxf(m, z);
    }
    red[tid] = m; __syncthreads();
    for (int s = 128; s > 0; s >>= 1) {
        if (tid < s) red[tid] = fmaxf(red[tid], red[tid + s]);
        __syncthreads();
    }
    const float rowmax = red[0]; __syncthreads();

    // pass 2: sum exp
    float sm = 0.f;
    for (int jj = tid; jj < TAGS_N; jj += 256) {
        float z = fmaf(s0, Ws2[jj], fmaf(s1v, Ws2[TAGS_N + jj],
                  fmaf(s2, Ws2[2 * TAGS_N + jj],
                  fmaf(s3v, Ws2[3 * TAGS_N + jj], bs2[jj]))));
        sm += __expf(z - rowmax);
    }
    red[tid] = sm; __syncthreads();
    for (int s = 128; s > 0; s >>= 1) {
        if (tid < s) red[tid] += red[tid + s];
        __syncthreads();
    }
    const float inv = 1.f / red[0]; __syncthreads();

    // pass 3: write log(probs + 1e-12)
    float* op = outp + (size_t)t * TAGS_N;
    for (int jj = tid; jj < TAGS_N; jj += 256) {
        float z = fmaf(s0, Ws2[jj], fmaf(s1v, Ws2[TAGS_N + jj],
                  fmaf(s2, Ws2[2 * TAGS_N + jj],
                  fmaf(s3v, Ws2[3 * TAGS_N + jj], bs2[jj]))));
        float p = __expf(z - rowmax) * inv;
        op[jj] = __logf(p + 1e-12f);
    }
}

// ---------------------------------------------------------------------------
extern "C" void kernel_launch(void* const* d_in, const int* in_sizes, int n_in,
                              void* d_out, int out_size, void* d_ws, size_t ws_size,
                              hipStream_t stream) {
    const int*   sentence = (const int*)  d_in[0];
    const float* emb      = (const float*)d_in[1];
    const float* Wf  = (const float*)d_in[2];  const float* bf_ = (const float*)d_in[3];
    const float* Wi  = (const float*)d_in[4];  const float* bi  = (const float*)d_in[5];
    const float* Wg  = (const float*)d_in[6];  const float* bg  = (const float*)d_in[7];
    const float* Wo  = (const float*)d_in[8];  const float* bo  = (const float*)d_in[9];
    const float* Wtag = (const float*)d_in[10]; const float* btag = (const float*)d_in[11];
    const float* Ws1  = (const float*)d_in[12]; const float* bs1  = (const float*)d_in[13];
    const float* Ws2  = (const float*)d_in[14]; const float* bs2  = (const float*)d_in[15];
    float* out = (float*)d_out;

    // workspace carve-up (~160.1 MB)
    char* ws = (char*)d_ws;
    size_t o = 0;
    unsigned short* W4B   = (unsigned short*)(ws + o); o += (size_t)E_DIM * G4 * 2;      // 8 MB
    unsigned short* Wh4T  = (unsigned short*)(ws + o); o += (size_t)G4 * H_DIM * 2;      // 8 MB
    unsigned short* embB  = (unsigned short*)(ws + o); o += (size_t)T_STEPS * E_DIM * 2; // 8 MB
    unsigned short* WtagB = (unsigned short*)(ws + o); o += (size_t)H_DIM * TAGS_N * 2;  // 64 MB
    float*          X4    = (float*)(ws + o);          o += (size_t)T_STEPS * G4 * 4;    // 64 MB
    unsigned short* hseqB = (unsigned short*)(ws + o); o += (size_t)T_STEPS * H_DIM * 2; // 8 MB
    float*          b4    = (float*)(ws + o);          o += (size_t)G4 * 4;
    float*          hbuf  = (float*)(ws + o);          o += (size_t)H_DIM * 4;
    float*          s1b   = (float*)(ws + o);          o += (size_t)T_STEPS * 4 * 4;
    unsigned int*   bar   = (unsigned int*)(ws + o);   o += 256;

    // phase 0: init + operand prep (bf16 conversion / transposes / gather)
    k_init<<<1, 256, 0, stream>>>(hbuf, bar);
    k_build_w4b <<<(E_DIM * G4) / 256, 256, 0, stream>>>(Wf, Wi, Wg, Wo, W4B);
    k_build_wh4t<<<(G4 * H_DIM) / 256, 256, 0, stream>>>(Wf, Wi, Wg, Wo, Wh4T);
    k_build_b4  <<<G4 / 256, 256, 0, stream>>>(bf_, bi, bg, bo, b4);
    k_gather_emb<<<(T_STEPS * E_DIM) / 256, 256, 0, stream>>>(sentence, emb, embB);
    k_conv_bf16 <<<(H_DIM * TAGS_N) / 256, 256, 0, stream>>>(Wtag, WtagB);

    // phase 1 (WMMA): X4 = embB @ W4B + b4   [4096 x 4096], hoisted x-part of gates
    k_gemm_bf16<<<dim3(G4 / 64, T_STEPS / 128), 256, 0, stream>>>(
        embB, W4B, X4, b4, E_DIM, E_DIM, G4, G4);

    // phase 2: sequential recurrence (L2-resident bf16 Wh, persistent grid)
    k_lstm<<<LSTM_WG, 256, 0, stream>>>(Wh4T, X4, hbuf, hseqB, bar);

    // phase 3 (WMMA): tag_logits = hseqB @ WtagB + b_tag -> d_out (scratch reuse)
    k_gemm_bf16<<<dim3(TAGS_N / 64, T_STEPS / 128), 256, 0, stream>>>(
        hseqB, WtagB, out, btag, H_DIM, H_DIM, TAGS_N, TAGS_N);

    // phase 4: rank-4 sampler + log-softmax (overwrites d_out with final result)
    k_samp1<<<T_STEPS, 256, 0, stream>>>(out, Ws1, bs1, s1b);
    k_samp2<<<T_STEPS, 256, 0, stream>>>(s1b, Ws2, bs2, out);
}
